// PPAPEMA_38027640438847
// MI455X (gfx1250) — matrix operand order repaired and checked by reading, not compiled, not run on hardware
//
#include <hip/hip_runtime.h>
#include <hip/hip_bf16.h>

#define G_    8
#define CG    32
#define HH    80
#define WW    80
#define HW    (HH*WW)        // 6400
#define SLICE (CG*HW)        // 204800
#define BG    128            // B*G

typedef __attribute__((ext_vector_type(16))) __bf16 bf16x16;
typedef __attribute__((ext_vector_type(8)))  __bf16 bf16x8;
typedef __attribute__((ext_vector_type(8)))  float  f32x8;
typedef __attribute__((ext_vector_type(4))) unsigned int u32x4;
typedef __attribute__((ext_vector_type(8))) int i32x8;
typedef __attribute__((ext_vector_type(4))) int i32x4;

// ---------------------------------------------------------------------------
// K1: axial pools -> 1x1 channel mix -> sigmoid gates; then GroupNorm stats
// of x_ax = x * a_w * a_h, computed on the fly. One block per bg slice.
// (Slice is 800KB -> passes 2 and 3 hit L2.)
// ---------------------------------------------------------------------------
__global__ __launch_bounds__(1024)
void k1_axial(const float* __restrict__ x, const float* __restrict__ axw,
              const float* __restrict__ axb,
              float* __restrict__ aw, float* __restrict__ ah,
              float* __restrict__ gn /*[BG][2] = mu, rstd*/)
{
    __shared__ float poolw[CG*WW];
    __shared__ float poolh[CG*HH];
    __shared__ float awl[CG*WW];
    __shared__ float ahl[CG*HH];
    const int bg  = blockIdx.x;
    const int tid = threadIdx.x;
    const float* xs = x + (size_t)bg * SLICE;

    // phase 1: axial mean pools
    for (int t = tid; t < CG*WW; t += blockDim.x) {
        int c = t / WW, w = t % WW;
        float s = 0.f;
        const float* p = xs + c*HW + w;
        #pragma unroll 4
        for (int h = 0; h < HH; ++h) s += p[h*WW];
        poolw[t] = s * (1.0f/HH);
    }
    for (int t = tid; t < CG*HH; t += blockDim.x) {
        int c = t / HH, h = t % HH;
        float s = 0.f;
        const float* p = xs + c*HW + h*WW;
        #pragma unroll 4
        for (int w = 0; w < WW; ++w) s += p[w];
        poolh[t] = s * (1.0f/WW);
    }
    __syncthreads();

    // phase 2: 32x32 channel mix over length 160, + bias, sigmoid
    for (int t = tid; t < CG*(WW+HH); t += blockDim.x) {
        int o = t / (WW+HH), l = t % (WW+HH);
        float s = axb[o];
        if (l < WW) {
            #pragma unroll
            for (int c = 0; c < CG; ++c) s += axw[o*CG+c] * poolw[c*WW+l];
        } else {
            int h = l - WW;
            #pragma unroll
            for (int c = 0; c < CG; ++c) s += axw[o*CG+c] * poolh[c*HH+h];
        }
        float a = 1.f / (1.f + __expf(-s));
        if (l < WW) { awl[o*WW+l] = a; aw[(size_t)(bg*CG+o)*WW + l] = a; }
        else        { ahl[o*HH + (l-WW)] = a; ah[(size_t)(bg*CG+o)*HH + (l-WW)] = a; }
    }
    __syncthreads();

    // phase 3: GroupNorm(num_groups=1) sum / sumsq of x_ax
    float s1 = 0.f, s2 = 0.f;
    for (int t = tid; t < SLICE; t += blockDim.x) {
        int c = t / HW, rem = t % HW;
        int h = rem / WW, w = rem % WW;
        float v = xs[t] * awl[c*WW+w] * ahl[c*HH+h];
        s1 += v; s2 += v*v;
    }
    poolw[tid] = s1; poolh[tid] = s2;   // pools are dead now; reuse as scratch
    __syncthreads();
    for (int off = blockDim.x >> 1; off > 0; off >>= 1) {
        if (tid < off) { poolw[tid] += poolw[tid+off]; poolh[tid] += poolh[tid+off]; }
        __syncthreads();
    }
    if (tid == 0) {
        float mu  = poolw[0] * (1.0f/(float)SLICE);
        float var = poolh[0] * (1.0f/(float)SLICE) - mu*mu;
        gn[bg*2+0] = mu;
        gn[bg*2+1] = rsqrtf(var + 1e-5f);
    }
}

// ---------------------------------------------------------------------------
// K3: 3x3 SAME conv (32->32 per group) as implicit GEMM with bf16 WMMA.
// Block = (bg, 4-row band). LDS: xs[y][x][ci] channel-interleaved so each
// lane's K-slice is two 16B LDS reads. 8 waves x 5 tiles x 9 taps.
// Weights staged into LDS via the Tensor Data Mover (one 1-D descriptor),
// then reshuffled tap-major in bf16. Output stored bf16 (dot-product input).
// ---------------------------------------------------------------------------
__global__ __launch_bounds__(256)
void k3_conv(const float* __restrict__ x, const float* __restrict__ lw,
             __bf16* __restrict__ xloc)
{
    __shared__ __attribute__((aligned(64))) __bf16 xs[6*82*CG];   // [y][x][ci]
    __shared__ __attribute__((aligned(64))) __bf16 wl[9*CG*CG];   // [tap][co][ci]
    __shared__ __attribute__((aligned(64))) float  wstage[9*CG*CG]; // raw fp32 weights
    const int bg   = blockIdx.y;
    const int band = blockIdx.x;          // 0..19, 4 rows each
    const int row0 = band * 4;
    const int tid  = threadIdx.x;

#if __has_builtin(__builtin_amdgcn_tensor_load_to_lds) && __has_builtin(__builtin_amdgcn_s_wait_tensorcnt)
    // --- TDM: DMA the 36864B fp32 weight block into LDS (wave 0 issues) ---
    if (tid < 32) {
        unsigned long long ga = (unsigned long long)(const void*)lw;
        unsigned int laddr = (unsigned int)(unsigned long long)(void*)&wstage[0];
        u32x4 g0;
        g0[0] = 1u;                                            // count=1 (valid user D#)
        g0[1] = laddr;                                         // lds_addr
        g0[2] = (unsigned int)ga;                              // global_addr[31:0]
        g0[3] = (unsigned int)((ga >> 32) & 0x01FFFFFFull)     // global_addr[56:32]
              | (2u << 30);                                    // type = 2 ("image")
        i32x8 g1;
        g1[0] = (int)(2u << 16);                               // data_size = 4B
        g1[1] = (int)((9216u & 0xFFFFu) << 16);                // tensor_dim0[15:0]
        g1[2] = (int)((1u) << 16);                             // tensor_dim0 hi | tensor_dim1=1
        g1[3] = (int)((9216u & 0xFFFFu) << 16);                // tensor_dim1 hi | tile_dim0=9216
        g1[4] = 1;                                             // tile_dim1=1, tile_dim2=0
        g1[5] = 9216;                                          // tensor_dim0_stride lo
        g1[6] = (int)((9216u & 0xFFFFu) << 16);                // stride0 hi | stride1 lo16
        g1[7] = 0;                                             // stride1 hi
        i32x4 z4 = {0, 0, 0, 0};
        i32x8 z8 = {0, 0, 0, 0, 0, 0, 0, 0};
        __builtin_amdgcn_tensor_load_to_lds(g0, g1, z4, z4, z8, 0);
        __builtin_amdgcn_s_wait_tensorcnt(0);
    }
    __syncthreads();
    // reshuffle staged weights: [co][ci][dy][dx] -> wl[tap][co][ci] in bf16
    for (int t = tid; t < 9*CG*CG; t += 256) {
        int tap = t / (CG*CG), rem = t % (CG*CG);
        int co = rem / CG, ci = rem % CG;
        int dy = tap / 3, dx = tap % 3;
        wl[t] = (__bf16)wstage[((co*CG + ci)*3 + dy)*3 + dx];
    }
#else
    for (int t = tid; t < 9*CG*CG; t += 256) {
        int tap = t / (CG*CG), rem = t % (CG*CG);
        int co = rem / CG, ci = rem % CG;
        int dy = tap / 3, dx = tap % 3;
        wl[t] = (__bf16)lw[((co*CG + ci)*3 + dy)*3 + dx];
    }
#endif

    // input tile with halo, zero-padded, fp32 -> bf16
    const float* xg = x + (size_t)bg * SLICE;
    for (int t = tid; t < 6*82*CG; t += 256) {
        int c = t / (6*82), rem = t % (6*82);
        int y = rem / 82, xx = rem % 82;
        int gh = row0 - 1 + y, gw = xx - 1;
        float v = 0.f;
        if (gh >= 0 && gh < HH && gw >= 0 && gw < WW) v = xg[c*HW + gh*WW + gw];
        xs[(y*82 + xx)*CG + c] = (__bf16)v;
    }
    __syncthreads();

    const int wid    = tid >> 5;
    const int lane   = tid & 31;
    const int mtile  = wid >> 2;          // 0..1 (co 0..15 / 16..31)
    const int nbase  = (wid & 3) * 5;     // 5 n-tiles per wave, one band row
    const int lhalf  = (lane < 16) ? 0 : 1;
    const int n_lane = lane & 15;

    f32x8 acc[5] = {};
    #pragma unroll
    for (int tap = 0; tap < 9; ++tap) {
        const int dy = tap / 3, dx = tap % 3;
        // A tile (16x32 bf16): lane m = lane&15 (+mtile*16 channel offset)
        const int m = mtile*16 + n_lane;
        const int aoff = lhalf ? 8 : 0;
        const __bf16* wb = &wl[(tap*CG + m)*CG];
        bf16x8 alo = *(const bf16x8*)(wb + aoff);        // K = off..off+7
        bf16x8 ahi = *(const bf16x8*)(wb + 16 + aoff);   // K = 16+off..16+off+7
        bf16x16 A = __builtin_shufflevector(alo, ahi,
            0,1,2,3,4,5,6,7,8,9,10,11,12,13,14,15);
        #pragma unroll
        for (int q = 0; q < 5; ++q) {
            const int ntile = nbase + q;
            const int r   = ntile / 5;                    // row in band
            const int col = (ntile % 5)*16 + n_lane;      // output col
            // B tile (32x16 bf16): lanes 0-15 hold K=0..7,16..23; 16-31 hold K=8..15,24..31
            const __bf16* bb = &xs[((r + dy)*82 + (col + dx))*CG + (lhalf ? 8 : 0)];
            bf16x8 blo = *(const bf16x8*)(bb);
            bf16x8 bhi = *(const bf16x8*)(bb + 16);
            bf16x16 Bm = __builtin_shufflevector(blo, bhi,
                0,1,2,3,4,5,6,7,8,9,10,11,12,13,14,15);
            acc[q] = __builtin_amdgcn_wmma_f32_16x16x32_bf16(
                false, A, false, Bm, (short)0, acc[q], false, false);
        }
    }

    // C layout: vgpr r -> M = r + (lane>=16 ? 8 : 0); N = lane&15
    #pragma unroll
    for (int q = 0; q < 5; ++q) {
        const int ntile = nbase + q;
        const int r   = ntile / 5;
        const int col = (ntile % 5)*16 + n_lane;
        const int gh  = row0 + r;
        #pragma unroll
        for (int v = 0; v < 8; ++v) {
            const int co = mtile*16 + v + (lhalf ? 8 : 0);
            xloc[((size_t)(bg*CG + co)*HH + gh)*WW + col] = (__bf16)acc[q][v];
        }
    }
}

// ---------------------------------------------------------------------------
// K4: PPAP top-64 mean per (bg, c) channel row, for x_ax (recomputed) and
// x_loc. Row cached in LDS; top-64 via threshold bisection.
// ---------------------------------------------------------------------------
__global__ __launch_bounds__(256)
void k4_topk(const float* __restrict__ x, const __bf16* __restrict__ xloc,
             const float* __restrict__ aw, const float* __restrict__ ah,
             float* __restrict__ ppax, float* __restrict__ pploc)
{
    __shared__ float row[HW];
    __shared__ float redf[256];
    __shared__ int   redi[256];
    __shared__ float bounds[2];
    const int idx   = blockIdx.x;          // [0, 8192)
    const int isloc = idx >> 12;
    const int rc    = idx & 4095;
    const int bg    = rc >> 5, c = rc & 31;
    const int tid   = threadIdx.x;
    const size_t base = (size_t)(bg*CG + c) * HW;

    if (isloc) {
        for (int i = tid; i < HW; i += 256) row[i] = (float)xloc[base + i];
    } else {
        const float* awp = aw + (size_t)(bg*CG + c)*WW;
        const float* ahp = ah + (size_t)(bg*CG + c)*HH;
        for (int i = tid; i < HW; i += 256)
            row[i] = x[base + i] * awp[i % WW] * ahp[i / WW];
    }
    __syncthreads();

    float mx = -3.4e38f, mn = 3.4e38f;
    for (int i = tid; i < HW; i += 256) { float v = row[i]; mx = fmaxf(mx, v); mn = fminf(mn, v); }
    redf[tid] = mx; __syncthreads();
    for (int o = 128; o > 0; o >>= 1) { if (tid < o) redf[tid] = fmaxf(redf[tid], redf[tid+o]); __syncthreads(); }
    if (tid == 0) bounds[1] = redf[0];
    __syncthreads();
    redf[tid] = mn; __syncthreads();
    for (int o = 128; o > 0; o >>= 1) { if (tid < o) redf[tid] = fminf(redf[tid], redf[tid+o]); __syncthreads(); }
    if (tid == 0) bounds[0] = redf[0];
    __syncthreads();

    float lo = bounds[0], hi = bounds[1];
    for (int it = 0; it < 32; ++it) {
        float mid = 0.5f * (lo + hi);
        int cnt = 0;
        for (int i = tid; i < HW; i += 256) cnt += (row[i] > mid) ? 1 : 0;
        redi[tid] = cnt; __syncthreads();
        for (int o = 128; o > 0; o >>= 1) { if (tid < o) redi[tid] += redi[tid+o]; __syncthreads(); }
        int total = redi[0];
        __syncthreads();
        if (total >= 64) lo = mid; else hi = mid;   // invariant: count(>hi) < 64
    }

    float s = 0.f; int cnt = 0;
    for (int i = tid; i < HW; i += 256) { float v = row[i]; if (v > hi) { s += v; ++cnt; } }
    redf[tid] = s; redi[tid] = cnt; __syncthreads();
    for (int o = 128; o > 0; o >>= 1) {
        if (tid < o) { redf[tid] += redf[tid+o]; redi[tid] += redi[tid+o]; }
        __syncthreads();
    }
    if (tid == 0) {
        float tsum = redf[0] + (float)(64 - redi[0]) * hi;  // pad with kth value
        float m = tsum * (1.0f/64.0f);
        if (isloc) pploc[rc] = m; else ppax[rc] = m;
    }
}

// ---------------------------------------------------------------------------
// K4b: fold GN affine into PPAP(ax), channel softmaxes, precompute
// qs = q_loc * s and qconst = sum(q_loc * t). One wave per bg.
// ---------------------------------------------------------------------------
__global__ __launch_bounds__(32)
void k4b_softmax(const float* __restrict__ gn, const float* __restrict__ gnw,
                 const float* __restrict__ gnb,
                 const float* __restrict__ ppax, const float* __restrict__ pploc,
                 float* __restrict__ qax, float* __restrict__ qs,
                 float* __restrict__ qconst)
{
    __shared__ float sh[CG];
    __shared__ float sh2[CG];
    const int bg = blockIdx.x, c = threadIdx.x;
    const float mu = gn[bg*2], rstd = gn[bg*2+1];
    const float s = rstd * gnw[c];
    const float t = gnb[c] - mu * s;
    const float vax  = s * ppax[bg*CG + c] + t;   // PPAP commutes with positive affine
    const float vloc = pploc[bg*CG + c];

    sh[c] = vax; __syncthreads();
    float m = -3.4e38f;
    #pragma unroll
    for (int i = 0; i < CG; ++i) m = fmaxf(m, sh[i]);
    float eax = __expf(vax - m);
    sh2[c] = eax; __syncthreads();
    float sum = 0.f;
    #pragma unroll
    for (int i = 0; i < CG; ++i) sum += sh2[i];
    float qa = eax / sum;
    __syncthreads();

    sh[c] = vloc; __syncthreads();
    m = -3.4e38f;
    #pragma unroll
    for (int i = 0; i < CG; ++i) m = fmaxf(m, sh[i]);
    float el = __expf(vloc - m);
    sh2[c] = el; __syncthreads();
    sum = 0.f;
    #pragma unroll
    for (int i = 0; i < CG; ++i) sum += sh2[i];
    float ql = el / sum;

    qax[bg*CG + c] = qa;
    qs[bg*CG + c]  = ql * s;
    __syncthreads();
    sh[c] = ql * t; __syncthreads();
    if (c == 0) {
        float qc = 0.f;
        #pragma unroll
        for (int i = 0; i < CG; ++i) qc += sh[i];
        qconst[bg] = qc;
    }
}

// ---------------------------------------------------------------------------
// K5: fused cross-attention + output. Per pixel: channel dots with q_ax
// (vs x_loc) and qs (vs recomputed x_ax), sigmoid, y = x_ax * attn.
// x_ax kept in registers between the dot pass and the write pass.
// ---------------------------------------------------------------------------
__global__ __launch_bounds__(256)
void k5_final(const float* __restrict__ x, const __bf16* __restrict__ xloc,
              const float* __restrict__ aw, const float* __restrict__ ah,
              const float* __restrict__ qax, const float* __restrict__ qs,
              const float* __restrict__ qconst, float* __restrict__ y)
{
    __shared__ float qa_s[CG], qs_s[CG];
    const int bg  = blockIdx.y;
    const int tid = threadIdx.x;
    const int l   = blockIdx.x * 256 + tid;      // pixel index [0, 6400)
    if (tid < CG) { qa_s[tid] = qax[bg*CG + tid]; qs_s[tid] = qs[bg*CG + tid]; }
    __syncthreads();

    const int w = l % WW, h = l / WW;
    const size_t base = (size_t)bg * SLICE + l;
    float acc = qconst[bg];
    float xa[CG];
    #pragma unroll
    for (int c = 0; c < CG; ++c) {
        float xv = x[base + (size_t)c * HW];
        float av = aw[(bg*CG + c)*WW + w] * ah[(bg*CG + c)*HH + h];
        float xl = (float)xloc[base + (size_t)c * HW];
        float va = xv * av;
        xa[c] = va;
        acc += qa_s[c] * xl + qs_s[c] * va;
    }
    const float attn = 1.f / (1.f + __expf(-acc));
    #pragma unroll
    for (int c = 0; c < CG; ++c)
        y[base + (size_t)c * HW] = xa[c] * attn;
}

// ---------------------------------------------------------------------------
extern "C" void kernel_launch(void* const* d_in, const int* in_sizes, int n_in,
                              void* d_out, int out_size, void* d_ws, size_t ws_size,
                              hipStream_t stream) {
    const float* x   = (const float*)d_in[0];
    const float* lw  = (const float*)d_in[1];
    const float* axw = (const float*)d_in[2];
    const float* axb = (const float*)d_in[3];
    const float* gnw = (const float*)d_in[4];
    const float* gnb = (const float*)d_in[5];
    float* out = (float*)d_out;
    float* ws  = (float*)d_ws;

    // workspace layout (float offsets)
    const size_t AW_OFF  = 0;                    // 128*32*80 = 327680
    const size_t AH_OFF  = 327680;               // 327680
    const size_t GN_OFF  = 655360;               // 256
    const size_t PAX_OFF = 655616;               // 4096
    const size_t PLC_OFF = 659712;               // 4096
    const size_t QAX_OFF = 663808;               // 4096
    const size_t QS_OFF  = 667904;               // 4096
    const size_t QC_OFF  = 672000;               // 128
    const size_t XL_OFF  = 1u << 20;             // x_loc: 26214400 bf16

    float*  aw   = ws + AW_OFF;
    float*  ah   = ws + AH_OFF;
    float*  gn   = ws + GN_OFF;
    float*  ppax = ws + PAX_OFF;
    float*  pplc = ws + PLC_OFF;
    float*  qax  = ws + QAX_OFF;
    float*  qs   = ws + QS_OFF;
    float*  qc   = ws + QC_OFF;
    __bf16* xl   = (__bf16*)(ws + XL_OFF);

    k1_axial<<<BG, 1024, 0, stream>>>(x, axw, axb, aw, ah, gn);
    k3_conv<<<dim3(20, BG), 256, 0, stream>>>(x, lw, xl);
    k4_topk<<<2*BG*CG, 256, 0, stream>>>(x, xl, aw, ah, ppax, pplc);
    k4b_softmax<<<BG, 32, 0, stream>>>(gn, gnw, gnb, ppax, pplc, qax, qs, qc);
    k5_final<<<dim3(HW/256, BG), 256, 0, stream>>>(x, xl, aw, ah, qax, qs, qc, out);
}
